// Model_29626684407935
// MI455X (gfx1250) — matrix operand order repaired and checked by reading, not compiled, not run on hardware
//
#include <hip/hip_runtime.h>
#include <math.h>

// ---------------------------------------------------------------------------
// BiLSTM(1->128) -> additive attention (HA=64) -> LSTM(256->128) -> Linear.
// B=8, L=512.  All matmuls: v_wmma_f32_16x16x32_f16, all operands f16 in
// memory (wide b128 fragment loads), f32 accumulation/state.  Native CDNA5
// transcendentals (v_tanh_f32 / v_exp_f32) on the hot elementwise paths.
// ---------------------------------------------------------------------------

typedef _Float16 v16h __attribute__((ext_vector_type(16)));
typedef _Float16 v8h  __attribute__((ext_vector_type(8)));
typedef float    v8f  __attribute__((ext_vector_type(8)));

#define WMMA_F16(a, b, c) \
  __builtin_amdgcn_wmma_f32_16x16x32_f16(false, (a), false, (b), (short)0, (c), false, false)

#define SHUF16(lo, hi) \
  __builtin_shufflevector((lo), (hi), 0, 1, 2, 3, 4, 5, 6, 7, 8, 9, 10, 11, 12, 13, 14, 15)

#ifndef __has_builtin
#define __has_builtin(x) 0
#endif
#if __has_builtin(__builtin_amdgcn_global_load_async_to_lds_b128)
#define HAVE_ASYNC_LDS 1
// exact parameter types per hipcc diagnostic: 'int __vector(4) __device__ *'
typedef int v4i_vs __attribute__((vector_size(16)));
typedef __attribute__((address_space(1))) v4i_vs* async_gptr;
typedef __attribute__((address_space(3))) v4i_vs* async_lptr;
#define ASYNC_LDS_B128(gp, lp) \
  __builtin_amdgcn_global_load_async_to_lds_b128( \
      (async_gptr)(uintptr_t)(gp), (async_lptr)(uintptr_t)(lp), 0, 0)
#else
#define HAVE_ASYNC_LDS 0
#endif
#if __has_builtin(__builtin_amdgcn_s_wait_asynccnt)
#define WAIT_ASYNC() __builtin_amdgcn_s_wait_asynccnt(0)
#else
#define WAIT_ASYNC() asm volatile("s_wait_asynccnt 0x0" ::: "memory")
#endif

// --- fast native transcendentals (guarded; safe fallbacks) ------------------
__device__ __forceinline__ float fast_tanh(float x) {
#if __has_builtin(__builtin_amdgcn_tanhf)
  return __builtin_amdgcn_tanhf(x);
#elif __has_builtin(__builtin_amdgcn_tanh_f32)
  return __builtin_amdgcn_tanh_f32(x);
#else
  return tanhf(x);
#endif
}
__device__ __forceinline__ float fast_exp(float x) {
#if __has_builtin(__builtin_amdgcn_exp2f)
  return __builtin_amdgcn_exp2f(x * 1.4426950408889634f);
#else
  return expf(x);
#endif
}
__device__ __forceinline__ float fast_rcp(float x) {
#if __has_builtin(__builtin_amdgcn_rcpf)
  return __builtin_amdgcn_rcpf(x);
#else
  return 1.f / x;
#endif
}
__device__ __forceinline__ float sigf(float v) { return fast_rcp(1.f + fast_exp(-v)); }

// --- f16 fragment loaders (two 16B vector loads each) -----------------------
// A fragment 16x32: lanes 0-15 M=lane (K base 0), lanes 16-31 K base +8;
// elements 0..7 -> K = kb+0..7, elements 8..15 -> K = kb+16..23.
__device__ __forceinline__ v16h load_a16_full(const _Float16* A, int lda, int m0,
                                              int k0, int lane) {
  const _Float16* p = A + (long)(m0 + (lane & 15)) * lda + k0 + ((lane >> 4) << 3);
  v8h lo = *(const v8h*)p;
  v8h hi = *(const v8h*)(p + 16);
  return SHUF16(lo, hi);
}

__device__ __forceinline__ v16h load_a16_masked(const _Float16* A, int lda, int m0,
                                                int k0, int M, int lane) {
  int m = m0 + (lane & 15);
  v16h r;
  if (m < M) {
    const _Float16* p = A + (long)m * lda + k0 + ((lane >> 4) << 3);
    v8h lo = *(const v8h*)p;
    v8h hi = *(const v8h*)(p + 16);
    r = SHUF16(lo, hi);
  } else {
#pragma unroll
    for (int e = 0; e < 16; ++e) r[e] = (_Float16)0.f;
  }
  return r;
}

// A fragment from LDS h-state (16 rows x 128, rows 8..15 pre-zeroed).
__device__ __forceinline__ v16h load_a_lds16(const _Float16* hl, int k0, int lane) {
  const _Float16* p = hl + (lane & 15) * 128 + k0 + ((lane >> 4) << 3);
  v8h lo = *(const v8h*)p;
  v8h hi = *(const v8h*)(p + 16);
  return SHUF16(lo, hi);
}

// B fragment 32x16 from N x K (row-major) f16 weights:
// lanes 0-15 -> N=lane, K=k0+0..7,16..23; lanes 16-31 -> K=k0+8..15,24..31.
__device__ __forceinline__ v16h load_b16_t(const _Float16* W, int ldw, int k0,
                                           int n0, int lane) {
  const _Float16* p = W + (long)(n0 + (lane & 15)) * ldw + k0 + ((lane >> 4) << 3);
  v8h lo = *(const v8h*)p;
  v8h hi = *(const v8h*)(p + 16);
  return SHUF16(lo, hi);
}

// --- generic batched f16 GEMM: C = A(MxK) * B^T(NxK) (+ col biases) ---------
__global__ void gemm16_kernel(const _Float16* __restrict__ A, int lda, long sA,
                              const _Float16* __restrict__ Bt, int ldb, long sB,
                              void* __restrict__ Cp, int ldc, long sC, int cfp16,
                              int M, int N, int K,
                              const float* __restrict__ bias0,
                              const float* __restrict__ bias1) {
  int batch = blockIdx.y;
  A  += (long)batch * sA;
  Bt += (long)batch * sB;

  int lane   = threadIdx.x & 31;
  int wave   = threadIdx.x >> 5;
  int ntiles = N >> 4;
  int mtiles = (M + 15) >> 4;
  int tile   = blockIdx.x * (blockDim.x >> 5) + wave;
  if (tile >= mtiles * ntiles) return;  // wave-uniform

  int m0 = (tile / ntiles) << 4;
  int n0 = (tile % ntiles) << 4;

  v8f acc = {};
  if (m0 + 16 <= M) {  // wave-uniform: common full-tile fast path, no masking
    for (int k0 = 0; k0 < K; k0 += 32) {
      v16h a = load_a16_full(A, lda, m0, k0, lane);
      v16h b = load_b16_t(Bt, ldb, k0, n0, lane);
      acc = WMMA_F16(a, b, acc);
    }
  } else {
    for (int k0 = 0; k0 < K; k0 += 32) {
      v16h a = load_a16_masked(A, lda, m0, k0, M, lane);
      v16h b = load_b16_t(Bt, ldb, k0, n0, lane);
      acc = WMMA_F16(a, b, acc);
    }
  }

  int n  = n0 + (lane & 15);
  int mb = m0 + ((lane >> 4) << 3);
  float bn = 0.f;
  if (bias0) bn += bias0[n];
  if (bias1) bn += bias1[n];
  if (cfp16) {
    _Float16* C = (_Float16*)Cp + (long)batch * sC;
#pragma unroll
    for (int v = 0; v < 8; ++v) {
      int m = mb + v;
      if (m < M) C[(long)m * ldc + n] = (_Float16)(acc[v] + bn);
    }
  } else {
    float* C = (float*)Cp + (long)batch * sC;
#pragma unroll
    for (int v = 0; v < 8; ++v) {
      int m = mb + v;
      if (m < M) C[(long)m * ldc + n] = acc[v] + bn;
    }
  }
}

// --- conversion helpers -----------------------------------------------------
__global__ void cvt16_kernel(const float* __restrict__ src,
                             _Float16* __restrict__ dst, long n) {
  long i = (long)blockIdx.x * blockDim.x + threadIdx.x;
  long stride = (long)gridDim.x * blockDim.x;
  for (; i < n; i += stride) dst[i] = (_Float16)src[i];
}

// transpose (RxC f32) -> (CxR f16)
__global__ void tcvt16_kernel(const float* __restrict__ src,
                              _Float16* __restrict__ dst, int R, int C) {
  int i = blockIdx.x * blockDim.x + threadIdx.x;
  if (i >= R * C) return;
  int r = i / C, c = i % C;
  dst[(long)c * R + r] = (_Float16)src[i];
}

// --- LSTM layer 1 (bidirectional; one 512-thread block per direction) -------
// Whh fragments live in registers; h-state in LDS (f16); c-state in registers.
__global__ void lstm1_kernel(const float* __restrict__ x,
                             const _Float16* __restrict__ whhf16,
                             const _Float16* __restrict__ whhb16,
                             const float* __restrict__ Wih_f, const float* __restrict__ bih_f,
                             const float* __restrict__ bhh_f,
                             const float* __restrict__ Wih_b, const float* __restrict__ bih_b,
                             const float* __restrict__ bhh_b,
                             _Float16* __restrict__ hcat16,
                             _Float16* __restrict__ hcatT16) {
  const int dir = blockIdx.x;
  const _Float16* whh = dir ? whhb16 : whhf16;
  const float* Wih = dir ? Wih_b : Wih_f;
  const float* bih = dir ? bih_b : bih_f;
  const float* bhh = dir ? bhh_b : bhh_f;

  __shared__ __align__(16) _Float16 h_lds[16 * 128];
  __shared__ float gbuf[8 * 512];

  int tid = threadIdx.x, lane = tid & 31, wave = tid >> 5;
#pragma unroll
  for (int r = 0; r < 4; ++r) h_lds[tid + (r << 9)] = (_Float16)0.f;
  float c0 = 0.f, c1 = 0.f;

  // per-wave N tiles: n0 = wave*32, wave*32+16 ; preload weight fragments
  v16h bfr0[4], bfr1[4];
#pragma unroll
  for (int kc = 0; kc < 4; ++kc) {
    bfr0[kc] = load_b16_t(whh, 128, kc * 32, wave * 32, lane);
    bfr1[kc] = load_b16_t(whh, 128, kc * 32, wave * 32 + 16, lane);
  }
  int gi0 = wave * 32 + (lane & 15);
  int gi1 = gi0 + 16;
  float wg0 = Wih[gi0], bs0 = bih[gi0] + bhh[gi0];
  float wg1 = Wih[gi1], bs1 = bih[gi1] + bhh[gi1];
  __syncthreads();

  for (int t = 0; t < 512; ++t) {
    int tt = dir ? (511 - t) : t;
    v8f acc0 = {}, acc1 = {};
#pragma unroll
    for (int kc = 0; kc < 4; ++kc) {
      v16h a = load_a_lds16(h_lds, kc * 32, lane);
      acc0 = WMMA_F16(a, bfr0[kc], acc0);
      acc1 = WMMA_F16(a, bfr1[kc], acc1);
    }
    int mb = (lane >> 4) << 3;
#pragma unroll
    for (int v = 0; v < 8; ++v) {
      int m = mb + v;
      if (m < 8) {
        float xv = x[m * 512 + tt];
        gbuf[m * 512 + gi0] = acc0[v] + xv * wg0 + bs0;
        gbuf[m * 512 + gi1] = acc1[v] + xv * wg1 + bs1;
      }
    }
    __syncthreads();
#pragma unroll
    for (int r = 0; r < 2; ++r) {
      int idx = tid + (r << 9);
      int b = idx >> 7, j = idx & 127;
      float ig = sigf(gbuf[b * 512 + j]);
      float fg = sigf(gbuf[b * 512 + 128 + j]);
      float gg = fast_tanh(gbuf[b * 512 + 256 + j]);
      float og = sigf(gbuf[b * 512 + 384 + j]);
      float& c = r ? c1 : c0;
      c = fg * c + ig * gg;
      float h = og * fast_tanh(c);
      _Float16 h16 = (_Float16)h;
      h_lds[idx] = h16;  // rows 0..7; rows 8..15 stay zero
      hcat16[((long)(b * 512 + tt)) * 256 + dir * 128 + j] = h16;
      hcatT16[((long)(b * 256 + dir * 128 + j)) * 512 + tt] = h16;
    }
    __syncthreads();
  }
}

// --- LSTM layer 2 (gate inputs precomputed in xg2, async-prefetched) --------
__global__ void lstm2_kernel(const float* __restrict__ xg2,
                             const _Float16* __restrict__ whh2_16,
                             _Float16* __restrict__ y16) {
  __shared__ __align__(16) _Float16 h_lds[16 * 128];
  __shared__ float gbuf[8 * 512];
#if HAVE_ASYNC_LDS
  __shared__ __align__(16) float xbuf[2][8 * 512];
#endif

  int tid = threadIdx.x, lane = tid & 31, wave = tid >> 5;
#pragma unroll
  for (int r = 0; r < 4; ++r) h_lds[tid + (r << 9)] = (_Float16)0.f;
  float c0 = 0.f, c1 = 0.f;

  v16h bfr0[4], bfr1[4];
#pragma unroll
  for (int kc = 0; kc < 4; ++kc) {
    bfr0[kc] = load_b16_t(whh2_16, 128, kc * 32, wave * 32, lane);
    bfr1[kc] = load_b16_t(whh2_16, 128, kc * 32, wave * 32 + 16, lane);
  }
  int gi0 = wave * 32 + (lane & 15);
  int gi1 = gi0 + 16;

#if HAVE_ASYNC_LDS
  // prime step 0: 8 rows x 512 f32 = 1024 x 16B chunks over 512 threads
#pragma unroll
  for (int i = 0; i < 2; ++i) {
    int cidx = tid + (i << 9);
    int m = cidx >> 7, col = (cidx & 127) << 2;
    const float* g = xg2 + ((long)(m * 512 + 0)) * 512 + col;
    float* l = &xbuf[0][cidx << 2];
    ASYNC_LDS_B128(g, l);
  }
  WAIT_ASYNC();
#endif
  __syncthreads();

  for (int t = 0; t < 512; ++t) {
#if HAVE_ASYNC_LDS
    if (t + 1 < 512) {  // prefetch next step while computing this one
#pragma unroll
      for (int i = 0; i < 2; ++i) {
        int cidx = tid + (i << 9);
        int m = cidx >> 7, col = (cidx & 127) << 2;
        const float* g = xg2 + ((long)(m * 512 + t + 1)) * 512 + col;
        float* l = &xbuf[(t + 1) & 1][cidx << 2];
        ASYNC_LDS_B128(g, l);
      }
    }
    const float* xb = xbuf[t & 1];
#endif
    v8f acc0 = {}, acc1 = {};
#pragma unroll
    for (int kc = 0; kc < 4; ++kc) {
      v16h a = load_a_lds16(h_lds, kc * 32, lane);
      acc0 = WMMA_F16(a, bfr0[kc], acc0);
      acc1 = WMMA_F16(a, bfr1[kc], acc1);
    }
    int mb = (lane >> 4) << 3;
#pragma unroll
    for (int v = 0; v < 8; ++v) {
      int m = mb + v;
      if (m < 8) {
#if HAVE_ASYNC_LDS
        gbuf[m * 512 + gi0] = acc0[v] + xb[m * 512 + gi0];
        gbuf[m * 512 + gi1] = acc1[v] + xb[m * 512 + gi1];
#else
        gbuf[m * 512 + gi0] = acc0[v] + xg2[((long)(m * 512 + t)) * 512 + gi0];
        gbuf[m * 512 + gi1] = acc1[v] + xg2[((long)(m * 512 + t)) * 512 + gi1];
#endif
      }
    }
    __syncthreads();
#pragma unroll
    for (int r = 0; r < 2; ++r) {
      int idx = tid + (r << 9);
      int b = idx >> 7, j = idx & 127;
      float ig = sigf(gbuf[b * 512 + j]);
      float fg = sigf(gbuf[b * 512 + 128 + j]);
      float gg = fast_tanh(gbuf[b * 512 + 256 + j]);
      float og = sigf(gbuf[b * 512 + 384 + j]);
      float& c = r ? c1 : c0;
      c = fg * c + ig * gg;
      float h = og * fast_tanh(c);
      h_lds[idx] = (_Float16)h;
      y16[((long)(b * 512 + t)) * 128 + j] = (_Float16)h;
    }
#if HAVE_ASYNC_LDS
    WAIT_ASYNC();
#endif
    __syncthreads();
  }
}

// --- attention scores: sig( Wa . tanh(q_t + k_s + bh) + ba ) ----------------
__global__ void attn_scores_kernel(const float* __restrict__ q,
                                   const float* __restrict__ kk,
                                   const float* __restrict__ bh,
                                   const float* __restrict__ Wa,
                                   const float* __restrict__ ba,
                                   float* __restrict__ scores) {
  int bt = blockIdx.x;  // b*512 + t
  int b  = bt >> 9;
  __shared__ float qrow[64];
  __shared__ float wa[64];
  int tid = threadIdx.x;
  if (tid < 64) { qrow[tid] = q[bt * 64 + tid] + bh[tid]; wa[tid] = Wa[tid]; }
  __syncthreads();
  float bav = ba[0];
  for (int s = tid; s < 512; s += 256) {
    const float* krow = kk + ((long)(b * 512 + s)) * 64;
    float e = 0.f;
#pragma unroll 8
    for (int h = 0; h < 64; ++h) e += wa[h] * fast_tanh(qrow[h] + krow[h]);
    scores[(long)bt * 512 + s] = sigf(e + bav);
  }
}

// --- softmax over last axis (512); output directly as f16 -------------------
__global__ void attn_softmax_kernel(const float* __restrict__ scores,
                                    _Float16* __restrict__ a16) {
  int bt = blockIdx.x;
  const float* row = scores + (long)bt * 512;
  __shared__ float red[256];
  int tid = threadIdx.x;
  float a0 = row[tid], a1 = row[tid + 256];
  red[tid] = fmaxf(a0, a1);
  __syncthreads();
  for (int off = 128; off > 0; off >>= 1) {
    if (tid < off) red[tid] = fmaxf(red[tid], red[tid + off]);
    __syncthreads();
  }
  float m = red[0];
  __syncthreads();
  float e0 = fast_exp(a0 - m), e1 = fast_exp(a1 - m);
  red[tid] = e0 + e1;
  __syncthreads();
  for (int off = 128; off > 0; off >>= 1) {
    if (tid < off) red[tid] += red[tid + off];
    __syncthreads();
  }
  float inv = fast_rcp(red[0]);
  a16[(long)bt * 512 + tid]       = (_Float16)(e0 * inv);
  a16[(long)bt * 512 + tid + 256] = (_Float16)(e1 * inv);
}

// ---------------------------------------------------------------------------
extern "C" void kernel_launch(void* const* d_in, const int* in_sizes, int n_in,
                              void* d_out, int out_size, void* d_ws, size_t ws_size,
                              hipStream_t stream) {
  (void)in_sizes; (void)n_in; (void)out_size; (void)ws_size;
  const float* x     = (const float*)d_in[0];
  const float* Wih_f = (const float*)d_in[1];
  const float* Whh_f = (const float*)d_in[2];
  const float* bih_f = (const float*)d_in[3];
  const float* bhh_f = (const float*)d_in[4];
  const float* Wih_b = (const float*)d_in[5];
  const float* Whh_b = (const float*)d_in[6];
  const float* bih_b = (const float*)d_in[7];
  const float* bhh_b = (const float*)d_in[8];
  const float* Wt    = (const float*)d_in[9];
  const float* Wx    = (const float*)d_in[10];
  const float* bh_a  = (const float*)d_in[11];
  const float* Wa    = (const float*)d_in[12];
  const float* ba    = (const float*)d_in[13];
  const float* Wih2  = (const float*)d_in[14];
  const float* Whh2  = (const float*)d_in[15];
  const float* bih2  = (const float*)d_in[16];
  const float* bhh2  = (const float*)d_in[17];
  const float* Wlin  = (const float*)d_in[18];
  const float* blin  = (const float*)d_in[19];
  float* out = (float*)d_out;

  // workspace layout (bytes)
  char* w = (char*)d_ws;
  _Float16* hcat16  = (_Float16*)(w);              // 4096x256        2 MB
  _Float16* hcatT16 = (_Float16*)(w + 2097152);    // 8 x 256x512     2 MB
  float*    xg2     = (float*)   (w + 4194304);    // 4096x512 f32    8 MB
  _Float16* y16     = (_Float16*)(w + 12582912);   // 4096x128        1 MB
  _Float16* ctx16   = (_Float16*)(w + 13631488);   // 4096x256        2 MB
  _Float16* whhf16  = (_Float16*)(w + 15728640);   // 512x128
  _Float16* whhb16  = (_Float16*)(w + 15859712);   // 512x128
  _Float16* whh2_16 = (_Float16*)(w + 15990784);   // 512x128
  _Float16* wtT16   = (_Float16*)(w + 16121856);   // 64x256
  _Float16* wxT16   = (_Float16*)(w + 16154624);   // 64x256
  _Float16* wih2_16 = (_Float16*)(w + 16187392);   // 512x256
  char*     pool    = w + 16449536;                // 16 MB pool
  float*    q       = (float*)(pool);              // 4096x64 f32     1 MB
  float*    kk      = (float*)(pool + 1048576);    // 4096x64 f32     1 MB
  float*    sc      = (float*)(pool + 2097152);    // 8x512x512 f32   8 MB
  _Float16* a16     = (_Float16*)(pool + 10485760);// 8x512x512 f16   4 MB
  _Float16* wlin16  = (_Float16*)(pool);           // 128x65536, after attn

  // 0) one-time weight conversions to f16 (Whh, Wih2 are N x K; Wt, Wx transposed to N x K)
  cvt16_kernel<<<dim3(64), dim3(256), 0, stream>>>(Whh_f, whhf16, 512L * 128);
  cvt16_kernel<<<dim3(64), dim3(256), 0, stream>>>(Whh_b, whhb16, 512L * 128);
  cvt16_kernel<<<dim3(64), dim3(256), 0, stream>>>(Whh2, whh2_16, 512L * 128);
  cvt16_kernel<<<dim3(128), dim3(256), 0, stream>>>(Wih2, wih2_16, 512L * 256);
  tcvt16_kernel<<<dim3(64), dim3(256), 0, stream>>>(Wt, wtT16, 256, 64);
  tcvt16_kernel<<<dim3(64), dim3(256), 0, stream>>>(Wx, wxT16, 256, 64);

  // 1) bidirectional LSTM -> hcat16 (B*L,256) and hcatT16 (B,256,512)
  lstm1_kernel<<<dim3(2), dim3(512), 0, stream>>>(
      x, whhf16, whhb16, Wih_f, bih_f, bhh_f, Wih_b, bih_b, bhh_b,
      hcat16, hcatT16);

  // 2) attention projections (M=4096,N=64,K=256), f32 outputs
  gemm16_kernel<<<dim3(128, 1), dim3(256), 0, stream>>>(
      hcat16, 256, 0, wtT16, 256, 0, q, 64, 0, 0, 4096, 64, 256, nullptr, nullptr);
  gemm16_kernel<<<dim3(128, 1), dim3(256), 0, stream>>>(
      hcat16, 256, 0, wxT16, 256, 0, kk, 64, 0, 0, 4096, 64, 256, nullptr, nullptr);

  // 3) scores + softmax (softmax emits f16 attention matrix)
  attn_scores_kernel<<<dim3(4096), dim3(256), 0, stream>>>(q, kk, bh_a, Wa, ba, sc);
  attn_softmax_kernel<<<dim3(4096), dim3(256), 0, stream>>>(sc, a16);

  // 4) context = a @ hcat, batched (M=512,N=256,K=512), f16 output
  gemm16_kernel<<<dim3(64, 8), dim3(256), 0, stream>>>(
      a16, 512, 512L * 512, hcatT16, 512, 256L * 512,
      ctx16, 256, 512L * 256, 1, 512, 256, 512, nullptr, nullptr);

  // 5) LSTM2 gate inputs: xg2 = ctx @ Wih2^T + bih2 + bhh2 (f32 out)
  gemm16_kernel<<<dim3(1024, 1), dim3(256), 0, stream>>>(
      ctx16, 256, 0, wih2_16, 256, 0, xg2, 512, 0, 0,
      4096, 512, 256, bih2, bhh2);

  // 6) convert Wlin (pool region now dead) then LSTM2 scan -> y16
  cvt16_kernel<<<dim3(1024), dim3(256), 0, stream>>>(Wlin, wlin16, 128L * 65536);
  lstm2_kernel<<<dim3(1), dim3(512), 0, stream>>>(xg2, whh2_16, y16);

  // 7) final linear: out = y @ Wlin^T + blin (M=8,N=128,K=65536)
  gemm16_kernel<<<dim3(1, 1), dim3(256), 0, stream>>>(
      y16, 65536, 0, wlin16, 65536, 0, out, 128, 0, 0,
      8, 128, 65536, blin, nullptr);
}
